// CrossAttention_71038759076322
// MI455X (gfx1250) — matrix operand order repaired and checked by reading, not compiled, not run on hardware
//
#include <hip/hip_runtime.h>
#include <hip/hip_bf16.h>
#include <math.h>

typedef __attribute__((ext_vector_type(16))) _Float16 v16h;
typedef __attribute__((ext_vector_type(8)))  _Float16 v8h;
typedef __attribute__((ext_vector_type(8)))  float    v8f;
typedef __attribute__((ext_vector_type(4)))  float    v4f;

// ---------------------------------------------------------------------------
// WMMA helper: D = A(16x32 f16) * B(32x16 f16) + C(16x16 f32)
// ---------------------------------------------------------------------------
__device__ __forceinline__ v8f wmma_f16(v16h a, v16h b, v8f c) {
    return __builtin_amdgcn_wmma_f32_16x16x32_f16(
        /*neg_a=*/false, a, /*neg_b=*/false, b,
        /*c_mod=*/(short)0, c, /*reuse_a=*/false, /*reuse_b=*/false);
}

// A-operand fragment (16x32, f16), row-major source with given row stride.
// ISA layout: lane m (m=lane&15, hi=lane>>4): halves 0..7 = K(hi*8 .. hi*8+7),
// halves 8..15 = K(16+hi*8 .. 16+hi*8+7).  Two b128 loads.
__device__ __forceinline__ v16h load_fragA(const _Float16* __restrict__ p,
                                           int stride, int lane) {
    int r  = lane & 15;
    int hi = lane >> 4;
    const _Float16* rp = p + r * stride + hi * 8;
    v8h lo = *(const v8h*)(rp);
    v8h hv = *(const v8h*)(rp + 16);
    v16h out;
#pragma unroll
    for (int i = 0; i < 8; ++i) { out[i] = lo[i]; out[i + 8] = hv[i]; }
    return out;
}

// B-operand fragment (32x16, f16).  Source holds B^T row-major: row index is
// the B column n, contiguous along K.  Layout matches A: lane n (g=lane>>4):
// halves 0..7 = K(g*8 .. g*8+7), halves 8..15 = K(16+g*8 .. 16+g*8+7).  Two b128 loads.
__device__ __forceinline__ v16h load_fragB(const _Float16* __restrict__ p,
                                           int stride, int lane) {
    int c = lane & 15;
    int g = lane >> 4;
    const _Float16* rp = p + c * stride + g * 8;
    v8h lo = *(const v8h*)(rp);
    v8h hv = *(const v8h*)(rp + 16);
    v16h out;
#pragma unroll
    for (int i = 0; i < 8; ++i) { out[i] = lo[i]; out[i + 8] = hv[i]; }
    return out;
}

// ---------------------------------------------------------------------------
// Stage 0: gather x/palette (b, C, 4096) into token-major f16 matrices using
// the torch-view-faithful token mapping r = b'*4096 + l,
// where s = b'*2048 + (l>>1), bb = l&1.
// ---------------------------------------------------------------------------
__global__ void ca_pack_tokens(const float* __restrict__ x,
                               const float* __restrict__ pal,
                               _Float16* __restrict__ Xt,
                               _Float16* __restrict__ Pt) {
    int idx = blockIdx.x * blockDim.x + threadIdx.x;   // e*8192 + r
    int e = idx >> 13;
    int r = idx & 8191;
    int b1 = r >> 12;
    int l  = r & 4095;
    int bb = l & 1;
    int s  = (b1 << 11) + (l >> 1);
    Xt[r * 256 + e] = (_Float16)x[((bb << 8) + e) * 4096 + s];
    if (e < 128)
        Pt[r * 128 + e] = (_Float16)pal[((bb << 7) + e) * 4096 + s];
}

__global__ void ca_cvt_f16(const float* __restrict__ in,
                           _Float16* __restrict__ out, int n) {
    int i = blockIdx.x * blockDim.x + threadIdx.x;
    if (i < n) out[i] = (_Float16)in[i];
}

// ---------------------------------------------------------------------------
// Stage 1: projection GEMM  C[r][o] = sum_e A[r][e]*W[o][e] + bias[o]
// One wave computes a 16(M) x 64(N) tile; Kin in steps of 32 (one WMMA).
// trans_out=1 writes C^T (used for V so the PV B-operand is contiguous).
// ---------------------------------------------------------------------------
__global__ __launch_bounds__(128) void ca_proj_gemm(
    const _Float16* __restrict__ A, const _Float16* __restrict__ W,
    const float* __restrict__ bias, _Float16* __restrict__ C,
    int Ntok, int Kin, int Nout, int trans_out) {
    int lane = threadIdx.x & 31;
    int wave = (blockIdx.x << 2) | (threadIdx.x >> 5);
    int ntn  = Nout >> 6;
    int mt = wave / ntn;
    int nt = wave % ntn;
    int r0 = mt << 4;
    int n0 = nt << 6;

    v8f acc[4] = {v8f{}, v8f{}, v8f{}, v8f{}};
    for (int kc = 0; kc < Kin; kc += 32) {
        v16h af = load_fragA(A + r0 * Kin + kc, Kin, lane);
#pragma unroll
        for (int t = 0; t < 4; ++t) {
            v16h bf = load_fragB(W + (n0 + t * 16) * Kin + kc, Kin, lane);
            acc[t] = wmma_f16(af, bf, acc[t]);
        }
    }

    int col = lane & 15, hi = lane >> 4;
#pragma unroll
    for (int t = 0; t < 4; ++t) {
        int o = n0 + t * 16 + col;
        float bv = bias[o];
        if (!trans_out) {
#pragma unroll
            for (int j = 0; j < 8; ++j) {
                int row = r0 + hi * 8 + j;
                C[row * Nout + o] = (_Float16)(acc[t][j] + bv);
            }
        } else {
            v8h pk;
#pragma unroll
            for (int j = 0; j < 8; ++j) pk[j] = (_Float16)(acc[t][j] + bv);
            *(v8h*)(C + (size_t)o * Ntok + r0 + hi * 8) = pk;
        }
    }
}

// ---------------------------------------------------------------------------
// Stage 2: flash attention.  Grid (16 bh-pairs, 64) x 128 threads; each wave
// owns one 16-row Q tile (256 tiles per bh pair), streams 64 keys/iteration:
// 4 WMMAs for S = Q K^T, online softmax, P via LDS relayout, 4 WMMAs for PV.
// ---------------------------------------------------------------------------
__global__ __launch_bounds__(128) void ca_flash_attn(
    const _Float16* __restrict__ Q, const _Float16* __restrict__ K,
    const _Float16* __restrict__ Vt, _Float16* __restrict__ O) {
    __shared__ _Float16 plds[4][16 * 64];   // per-wave 16x64 P tile (2 KB)

    int lane = threadIdx.x & 31;
    int wave = threadIdx.x >> 5;
    int bh = blockIdx.x;                     // 0..15
    int bp = bh >> 3;                        // batch half b'
    int hh = bh & 7;                         // head
    int qt = blockIdx.y * 4 + wave;          // 0..255
    int r0 = bp * 4096 + qt * 16;
    int hcol = hh * 32;
    const float scale = 0.1767766953f;       // 1/sqrt(32)

    v16h qf = load_fragA(Q + (size_t)r0 * 256 + hcol, 256, lane);

    v8f o0 = {}, o1 = {};                    // dims 0-15 / 16-31 accumulators
    float mrow[8], lrow[8];
#pragma unroll
    for (int j = 0; j < 8; ++j) { mrow[j] = -1.0e30f; lrow[j] = 0.0f; }

    _Float16* pl = &plds[wave][0];
    int colb = lane & 15;
    int rowb = (lane >> 4) * 8;

    for (int kb = 0; kb < 4096; kb += 64) {
        const _Float16* kbase = K + (size_t)(bp * 4096 + kb) * 256 + hcol;
        v8f s[4];
#pragma unroll
        for (int t = 0; t < 4; ++t) {
            v16h kf = load_fragB(kbase + t * 16 * 256, 256, lane);
            v8f z = {};
            s[t] = wmma_f16(qf, kf, z);
        }

        float p[4][8];
#pragma unroll
        for (int j = 0; j < 8; ++j) {
            float v0 = s[0][j] * scale, v1 = s[1][j] * scale;
            float v2 = s[2][j] * scale, v3 = s[3][j] * scale;
            float t = fmaxf(fmaxf(v0, v1), fmaxf(v2, v3));
            t = fmaxf(t, __shfl_xor(t, 1));
            t = fmaxf(t, __shfl_xor(t, 2));
            t = fmaxf(t, __shfl_xor(t, 4));
            t = fmaxf(t, __shfl_xor(t, 8));
            float mn = fmaxf(mrow[j], t);
            float corr = __expf(mrow[j] - mn);
            mrow[j] = mn;
            float p0 = __expf(v0 - mn), p1 = __expf(v1 - mn);
            float p2 = __expf(v2 - mn), p3 = __expf(v3 - mn);
            p[0][j] = p0; p[1][j] = p1; p[2][j] = p2; p[3][j] = p3;
            float sm = (p0 + p1) + (p2 + p3);
            sm += __shfl_xor(sm, 1);
            sm += __shfl_xor(sm, 2);
            sm += __shfl_xor(sm, 4);
            sm += __shfl_xor(sm, 8);
            lrow[j] = lrow[j] * corr + sm;
            o0[j] *= corr;
            o1[j] *= corr;
        }

        // D-fragment (row=hi*8+j, col=t*16+colb) -> LDS row-major 16x64
#pragma unroll
        for (int j = 0; j < 8; ++j) {
#pragma unroll
            for (int t = 0; t < 4; ++t)
                pl[(rowb + j) * 64 + t * 16 + colb] = (_Float16)p[t][j];
        }

        const _Float16* vbase = Vt + (size_t)hcol * 8192 + bp * 4096 + kb;
#pragma unroll
        for (int kk = 0; kk < 2; ++kk) {     // two 32-key sub-chunks
            v16h pf  = load_fragA(pl + kk * 32, 64, lane);
            v16h vf0 = load_fragB(vbase + kk * 32, 8192, lane);
            v16h vf1 = load_fragB(vbase + 16 * 8192 + kk * 32, 8192, lane);
            o0 = wmma_f16(pf, vf0, o0);
            o1 = wmma_f16(pf, vf1, o1);
        }
    }

    // epilogue: O = acc / l, store row-major f16 (8192 x 256)
#pragma unroll
    for (int j = 0; j < 8; ++j) {
        float inv = 1.0f / lrow[j];
        int row = r0 + rowb + j;
        O[(size_t)row * 256 + hcol + colb]      = (_Float16)(o0[j] * inv);
        O[(size_t)row * 256 + hcol + 16 + colb] = (_Float16)(o1[j] * inv);
    }
}

// ---------------------------------------------------------------------------
// Stage 3: out = x + O @ Wo^T + bo, written channel-major (b, 256, 4096).
// One wave per 16(M) x 64(N) tile; token index is contiguous per lane so the
// residual load / final store are b128 ops.
// ---------------------------------------------------------------------------
__global__ __launch_bounds__(128) void ca_out_proj(
    const _Float16* __restrict__ O, const _Float16* __restrict__ Wo16,
    const float* __restrict__ bo, const float* __restrict__ x,
    float* __restrict__ out) {
    int lane = threadIdx.x & 31;
    int wave = (blockIdx.x << 2) | (threadIdx.x >> 5);
    int mt = wave >> 2;
    int nt = wave & 3;
    int r0 = mt << 4;
    int n0 = nt << 6;

    v8f acc[4] = {v8f{}, v8f{}, v8f{}, v8f{}};
    for (int kc = 0; kc < 256; kc += 32) {
        v16h af = load_fragA(O + (size_t)r0 * 256 + kc, 256, lane);
#pragma unroll
        for (int t = 0; t < 4; ++t) {
            v16h bf = load_fragB(Wo16 + (n0 + t * 16) * 256 + kc, 256, lane);
            acc[t] = wmma_f16(af, bf, acc[t]);
        }
    }

    int col = lane & 15, hi = lane >> 4;
    int bp = r0 >> 12;
    int l0 = (r0 & 4095) + hi * 8;
#pragma unroll
    for (int t = 0; t < 4; ++t) {
        int ch = n0 + t * 16 + col;
        float bv = bo[ch];
        size_t base = (size_t)bp * 1048576 + (size_t)ch * 4096 + l0;
        v4f xa = *(const v4f*)(x + base);
        v4f xb = *(const v4f*)(x + base + 4);
        v4f ra, rb;
#pragma unroll
        for (int j = 0; j < 4; ++j) {
            ra[j] = xa[j] + bv + acc[t][j];
            rb[j] = xb[j] + bv + acc[t][4 + j];
        }
        *(v4f*)(out + base)     = ra;
        *(v4f*)(out + base + 4) = rb;
    }
}

// ---------------------------------------------------------------------------
extern "C" void kernel_launch(void* const* d_in, const int* in_sizes, int n_in,
                              void* d_out, int out_size, void* d_ws, size_t ws_size,
                              hipStream_t stream) {
    const float* x   = (const float*)d_in[0];
    const float* pal = (const float*)d_in[1];
    const float* Wq  = (const float*)d_in[2];
    const float* bq  = (const float*)d_in[3];
    const float* Wk  = (const float*)d_in[4];
    const float* bk  = (const float*)d_in[5];
    const float* Wv  = (const float*)d_in[6];
    const float* bv  = (const float*)d_in[7];
    const float* Wo  = (const float*)d_in[8];
    const float* bo  = (const float*)d_in[9];
    float* out = (float*)d_out;

    // workspace layout (f16 elements)
    _Float16* ws   = (_Float16*)d_ws;
    _Float16* Xt   = ws;                 // 8192*256
    _Float16* Pt   = Xt + 2097152;       // 8192*128
    _Float16* Wq16 = Pt + 1048576;       // 256*128
    _Float16* Wk16 = Wq16 + 32768;       // 256*256
    _Float16* Wv16 = Wk16 + 65536;
    _Float16* Wo16 = Wv16 + 65536;
    _Float16* Qm   = Wo16 + 65536;       // 8192*256
    _Float16* Km   = Qm + 2097152;       // 8192*256
    _Float16* Vt   = Km + 2097152;       // 256*8192 (transposed V)
    _Float16* Om   = Vt + 2097152;       // 8192*256

    ca_pack_tokens<<<8192, 256, 0, stream>>>(x, pal, Xt, Pt);
    ca_cvt_f16<<<(32768 + 255) / 256, 256, 0, stream>>>(Wq, Wq16, 32768);
    ca_cvt_f16<<<(65536 + 255) / 256, 256, 0, stream>>>(Wk, Wk16, 65536);
    ca_cvt_f16<<<(65536 + 255) / 256, 256, 0, stream>>>(Wv, Wv16, 65536);
    ca_cvt_f16<<<(65536 + 255) / 256, 256, 0, stream>>>(Wo, Wo16, 65536);

    // Q = Pt @ Wq^T + bq ; K = Xt @ Wk^T + bk ; V^T = (Xt @ Wv^T + bv)^T
    ca_proj_gemm<<<512, 128, 0, stream>>>(Pt, Wq16, bq, Qm, 8192, 128, 256, 0);
    ca_proj_gemm<<<512, 128, 0, stream>>>(Xt, Wk16, bk, Km, 8192, 256, 256, 0);
    ca_proj_gemm<<<512, 128, 0, stream>>>(Xt, Wv16, bv, Vt, 8192, 256, 256, 1);

    dim3 ag(16, 64);
    ca_flash_attn<<<ag, 128, 0, stream>>>(Qm, Km, Vt, Om);

    ca_out_proj<<<512, 128, 0, stream>>>(Om, Wo16, bo, x, out);
}
